// STGNN_77094662964037
// MI455X (gfx1250) — matrix operand order repaired and checked by reading, not compiled, not run on hardware
//
#include <hip/hip_runtime.h>
#include <hip/hip_bf16.h>
#include <cstdint>

typedef __attribute__((ext_vector_type(16))) _Float16 v16h;
typedef __attribute__((ext_vector_type(8)))  float    v8f;

#define WMMA_F16(a, b, c) \
  __builtin_amdgcn_wmma_f32_16x16x32_f16(false, (a), false, (b), (short)0, (c), false, false)

union FragH {
  v16h v;
  uint4 u[2];
  _Float16 f[16];
};

// B fragment (K x 16), weights stored [out][in] row-major with row stride `stride` halves.
// lane L: column n = n0 + L%16, elems 0-7 = K klo..klo+7, elems 8-15 = K klo+16..klo+23, klo = kc + 8*(L>=16)
__device__ inline v16h ldsLoadB(const _Float16* W, int stride, int n0, int kc, int lane) {
  int n = n0 + (lane & 15);
  int klo = kc + ((lane & 16) ? 8 : 0);
  const _Float16* p = W + n * stride + klo;
  FragH r;
  r.u[0] = *(const uint4*)(p);
  r.u[1] = *(const uint4*)(p + 16);
  return r.v;
}

// A fragment (16 x K) from LDS h stored [16][64] halves row-major.
// lane L: row m = L%16, elems 0-7 = K klo..klo+7, elems 8-15 = K klo+16..klo+23, klo = kc + 8*(L>=16)
__device__ inline v16h ldsLoadA(const _Float16* Hrow, int kc, int lane) {
  int m = lane & 15;
  int klo = kc + ((lane & 16) ? 8 : 0);
  const _Float16* p = Hrow + m * 64 + klo;
  FragH r;
  r.u[0] = *(const uint4*)(p);
  r.u[1] = *(const uint4*)(p + 16);
  return r.v;
}

__device__ inline float sigm(float x) { return 1.f / (1.f + __expf(-x)); }

// ---------------- one-time f16 weight packing into workspace ----------------
// layout (halves): [0,6144)=WihS  [6144,12288)=WihD(pad)  [12288,24576)=Whh  [24576,28672)=H1
__global__ __launch_bounds__(256) void k_wconv(const float* __restrict__ Wih,
                                               const float* __restrict__ Whh,
                                               const float* __restrict__ h1W,
                                               _Float16* __restrict__ dst) {
  int i = blockIdx.x * 256 + threadIdx.x;
  if (i < 6144) {
    int n = i >> 5, k = i & 31;
    dst[i] = (_Float16)Wih[n * 40 + k];
  } else if (i < 12288) {
    int j = i - 6144;
    int n = j >> 5, k = j & 31;
    dst[i] = (k < 8) ? (_Float16)Wih[n * 40 + 32 + k] : (_Float16)0.f;
  } else if (i < 24576) {
    dst[i] = (_Float16)Whh[i - 12288];
  } else if (i < 28672) {
    dst[i] = (_Float16)h1W[i - 24576];
  }
}

// ---------------- Graph phase ----------------

// Y[n][o] = b[o] + sum_k act(X[n][k]) * W[o][k]  (32x32 weights), also zeroes Z[n][o].
// Z may alias X (zero happens after staging + barrier; blocks own disjoint rows).
template <bool RELU>
__global__ __launch_bounds__(256) void k_linear(const float* __restrict__ X,
                                                const float* __restrict__ W,
                                                const float* __restrict__ b,
                                                float* __restrict__ Y,
                                                float* __restrict__ Z, int Nn) {
  __shared__ float sWt[32 * 32];  // transposed: sWt[k][o] to avoid bank conflicts
  __shared__ float sX[8 * 32];
  int tid = threadIdx.x;
#pragma unroll
  for (int i = 0; i < 4; ++i) {
    int idx = tid + 256 * i;
    int o = idx >> 5, k = idx & 31;
    sWt[k * 32 + o] = W[idx];
  }
  int nl = tid >> 5, o = tid & 31;
  int node = blockIdx.x * 8 + nl;
  {
    int srcNode = node < Nn ? node : Nn - 1;
    sX[tid] = X[(size_t)srcNode * 32 + o];
  }
  __syncthreads();
  if (node < Nn) {
    float acc = b[o];
#pragma unroll
    for (int k = 0; k < 32; ++k) {
      float x = sX[nl * 32 + k];
      if (RELU) x = fmaxf(x, 0.f);
      acc = fmaf(x, sWt[k * 32 + o], acc);
    }
    size_t off = (size_t)node * 32 + o;
    Y[off] = acc;
    Z[off] = 0.f;
  }
}

// scatter-add spmm: 32 lanes per edge (lane = feature), wave-uniform edge index
__global__ __launch_bounds__(256) void k_spmm(const int* __restrict__ rows,
                                              const int* __restrict__ cols,
                                              const float* __restrict__ vals,
                                              const float* __restrict__ XW,
                                              float* __restrict__ AGG, int Ee) {
  long long gid = (long long)blockIdx.x * 256 + threadIdx.x;
  int e = (int)(gid >> 5);
  int f = (int)(gid & 31);
  if (e >= Ee) return;
  int r = rows[e], c = cols[e];
  float v = vals[e];
  unsafeAtomicAdd(&AGG[(size_t)r * 32 + f], v * XW[(size_t)c * 32 + f]);
}

// ---------------- Fused GRU + head over all T steps ----------------
// Block = 2 waves, each wave owns 16 nodes for the whole time loop.
__global__ __launch_bounds__(64) void k_fused(
    const float* __restrict__ Xdyn, const float* __restrict__ hgIn,
    const uint* __restrict__ wpk,  // packed f16 weights (14336 dwords)
    const float* __restrict__ b_ih, const float* __restrict__ b_hh,
    const float* __restrict__ h1b, const float* __restrict__ h2W,
    const float* __restrict__ h2b, float* __restrict__ out, int Nn, int Tt) {
  __shared__ uint smem[15360];  // 61440 B: weights (14336 dw) + 2x h-state (512 dw each)

  const int tid = threadIdx.x;
  const int w = tid >> 5;
  const int lane = tid & 31;
  const int nodeBase = (blockIdx.x * 2 + w) * 16;

  _Float16* sWihS = (_Float16*)smem;       // 192x32
  _Float16* sWihD = sWihS + 6144;          // 192x32 (zero padded)
  _Float16* sWhh  = sWihS + 12288;         // 192x64
  _Float16* sH1   = sWihS + 24576;         // 64x64
  _Float16* hW    = sWihS + 28672 + w * 1024;  // per-wave 16x64

  for (int i = tid; i < 14336; i += 64) smem[i] = wpk[i];
  {
    uint* hU = (uint*)hW;
    for (int i = lane; i < 512; i += 32) hU[i] = 0u;
  }
  __syncthreads();

  // gi_static[j] = relu(h_g) @ W_ih[:, :32]^T tile j, C seeded with b_ih broadcast
  v8f giS[12];
  {
    FragH ahg;
    int m = lane & 15;
    int klo = (lane & 16) ? 8 : 0;
    int node = nodeBase + m;
    if (node >= Nn) node = Nn - 1;
    const float* hp = hgIn + (size_t)node * 32;
#pragma unroll
    for (int i = 0; i < 8; ++i) ahg.f[i] = (_Float16)fmaxf(hp[klo + i], 0.f);
#pragma unroll
    for (int i = 0; i < 8; ++i) ahg.f[8 + i] = (_Float16)fmaxf(hp[16 + klo + i], 0.f);
#pragma unroll
    for (int j = 0; j < 12; ++j) {
      float bb = b_ih[j * 16 + (lane & 15)];
      v8f c;
#pragma unroll
      for (int r = 0; r < 8; ++r) c[r] = bb;
      giS[j] = WMMA_F16(ahg.v, ldsLoadB(sWihS, 32, j * 16, 0, lane), c);
    }
  }

  // hoist loop-invariant per-lane scalars
  const float h2bias = h2b[0];
  float bhr[4], bhz[4], bhn[4], h1bb[4], h2w[4];
#pragma unroll
  for (int jj = 0; jj < 4; ++jj) {
    bhr[jj] = b_hh[(jj)*16 + (lane & 15)];
    bhz[jj] = b_hh[(jj + 4) * 16 + (lane & 15)];
    bhn[jj] = b_hh[(jj + 8) * 16 + (lane & 15)];
    h1bb[jj] = h1b[jj * 16 + (lane & 15)];
    h2w[jj] = h2W[jj * 16 + (lane & 15)];
  }

  for (int t = 0; t < Tt; ++t) {
    // x_dyn A fragment, zero-padded to K=32 (lanes>=16 cover K>=8 -> all zero)
    FragH ax;
    ax.u[0] = make_uint4(0, 0, 0, 0);
    ax.u[1] = make_uint4(0, 0, 0, 0);
    if (lane < 16) {
      int node = nodeBase + lane;
      if (node >= Nn) node = Nn - 1;
      const float4* xp = (const float4*)(Xdyn + ((size_t)t * Nn + node) * 8);
      float4 x0 = xp[0], x1 = xp[1];
      ax.f[0] = (_Float16)x0.x; ax.f[1] = (_Float16)x0.y;
      ax.f[2] = (_Float16)x0.z; ax.f[3] = (_Float16)x0.w;
      ax.f[4] = (_Float16)x1.x; ax.f[5] = (_Float16)x1.y;
      ax.f[6] = (_Float16)x1.z; ax.f[7] = (_Float16)x1.w;
    }
    v16h ah0 = ldsLoadA(hW, 0, lane);
    v16h ah1 = ldsLoadA(hW, 32, lane);

    float hnew[4][8];
#pragma unroll
    for (int jj = 0; jj < 4; ++jj) {
      // gi = gi_static + x_dyn @ W_ih_dyn^T   (gate tiles: r=jj, z=jj+4, n=jj+8)
      v8f gir = WMMA_F16(ax.v, ldsLoadB(sWihD, 32, (jj)*16, 0, lane), giS[jj]);
      v8f giz = WMMA_F16(ax.v, ldsLoadB(sWihD, 32, (jj + 4) * 16, 0, lane), giS[jj + 4]);
      v8f gin = WMMA_F16(ax.v, ldsLoadB(sWihD, 32, (jj + 8) * 16, 0, lane), giS[jj + 8]);
      // gh = h @ W_hh^T + b_hh
      v8f ghr, ghz, ghn;
#pragma unroll
      for (int r = 0; r < 8; ++r) { ghr[r] = bhr[jj]; ghz[r] = bhz[jj]; ghn[r] = bhn[jj]; }
      ghr = WMMA_F16(ah0, ldsLoadB(sWhh, 64, (jj)*16, 0, lane), ghr);
      ghr = WMMA_F16(ah1, ldsLoadB(sWhh, 64, (jj)*16, 32, lane), ghr);
      ghz = WMMA_F16(ah0, ldsLoadB(sWhh, 64, (jj + 4) * 16, 0, lane), ghz);
      ghz = WMMA_F16(ah1, ldsLoadB(sWhh, 64, (jj + 4) * 16, 32, lane), ghz);
      ghn = WMMA_F16(ah0, ldsLoadB(sWhh, 64, (jj + 8) * 16, 0, lane), ghn);
      ghn = WMMA_F16(ah1, ldsLoadB(sWhh, 64, (jj + 8) * 16, 32, lane), ghn);
#pragma unroll
      for (int r = 0; r < 8; ++r) {
        float rr = sigm(gir[r] + ghr[r]);
        float zz = sigm(giz[r] + ghz[r]);
        float nn = tanhf(gin[r] + rr * ghn[r]);
        int m = r + ((lane & 16) ? 8 : 0);
        float hp = (float)hW[m * 64 + jj * 16 + (lane & 15)];
        hnew[jj][r] = (1.f - zz) * nn + zz * hp;
      }
    }
    // commit h_new (all slices after full read of h_prev)
#pragma unroll
    for (int jj = 0; jj < 4; ++jj)
#pragma unroll
      for (int r = 0; r < 8; ++r) {
        int m = r + ((lane & 16) ? 8 : 0);
        hW[m * 64 + jj * 16 + (lane & 15)] = (_Float16)hnew[jj][r];
      }
    __syncthreads();

    // head: y = sigmoid(relu(h_new @ h1W^T + h1b) @ h2W^T + h2b)
    v16h ay0 = ldsLoadA(hW, 0, lane);
    v16h ay1 = ldsLoadA(hW, 32, lane);
    float p[8];
#pragma unroll
    for (int r = 0; r < 8; ++r) p[r] = 0.f;
#pragma unroll
    for (int jj = 0; jj < 4; ++jj) {
      v8f yc;
#pragma unroll
      for (int r = 0; r < 8; ++r) yc[r] = h1bb[jj];
      yc = WMMA_F16(ay0, ldsLoadB(sH1, 64, jj * 16, 0, lane), yc);
      yc = WMMA_F16(ay1, ldsLoadB(sH1, 64, jj * 16, 32, lane), yc);
#pragma unroll
      for (int r = 0; r < 8; ++r) p[r] += fmaxf(yc[r], 0.f) * h2w[jj];
    }
#pragma unroll
    for (int mask = 1; mask < 16; mask <<= 1)
#pragma unroll
      for (int r = 0; r < 8; ++r) p[r] += __shfl_xor(p[r], mask, 32);
    if ((lane & 15) == 0) {
      int mbase = (lane & 16) ? 8 : 0;
      int node0 = nodeBase + mbase;
      size_t o = (size_t)t * Nn + node0;
      if (node0 + 7 < Nn && ((o & 3) == 0)) {
        float4 a = make_float4(sigm(p[0] + h2bias), sigm(p[1] + h2bias),
                               sigm(p[2] + h2bias), sigm(p[3] + h2bias));
        float4 b4 = make_float4(sigm(p[4] + h2bias), sigm(p[5] + h2bias),
                                sigm(p[6] + h2bias), sigm(p[7] + h2bias));
        *(float4*)(out + o) = a;
        *(float4*)(out + o + 4) = b4;
      } else {
#pragma unroll
        for (int r = 0; r < 8; ++r)
          if (node0 + r < Nn) out[o + r] = sigm(p[r] + h2bias);
      }
    }
    __syncthreads();
  }
}

extern "C" void kernel_launch(void* const* d_in, const int* in_sizes, int n_in,
                              void* d_out, int out_size, void* d_ws, size_t ws_size,
                              hipStream_t stream) {
  const float* Xs  = (const float*)d_in[0];
  const float* Xd  = (const float*)d_in[1];
  const int*   er  = (const int*)d_in[2];
  const int*   ec  = (const int*)d_in[3];
  const float* ev  = (const float*)d_in[4];
  const float* g1W = (const float*)d_in[5];
  const float* g1b = (const float*)d_in[6];
  const float* g2W = (const float*)d_in[7];
  const float* g2b = (const float*)d_in[8];
  const float* Wih = (const float*)d_in[9];
  const float* Whh = (const float*)d_in[10];
  const float* bih = (const float*)d_in[11];
  const float* bhh = (const float*)d_in[12];
  const float* h1W = (const float*)d_in[13];
  const float* h1b = (const float*)d_in[14];
  const float* h2W = (const float*)d_in[15];
  const float* h2b = (const float*)d_in[16];
  float* out = (float*)d_out;

  int Nn = in_sizes[0] / 32;
  int Ee = in_sizes[2];
  int Tt = in_sizes[1] / (Nn * 8);

  float* bufA = (float*)d_ws;               // N x 32 f32
  float* bufB = bufA + (size_t)Nn * 32;     // N x 32 f32
  _Float16* wpk = (_Float16*)(bufB + (size_t)Nn * 32);  // 28672 halves packed f16

  int linBlocks = (Nn + 7) / 8;
  long long spThreads = (long long)Ee * 32;
  int spBlocks = (int)((spThreads + 255) / 256);
  int fusedBlocks = (Nn + 31) / 32;

  // pack f16 weights once (28672 halves / 256 = 112 blocks)
  k_wconv<<<112, 256, 0, stream>>>(Wih, Whh, h1W, wpk);
  // XW1 = Xs @ g1W^T + b ; zero AGG1(bufB)
  k_linear<false><<<linBlocks, 256, 0, stream>>>(Xs, g1W, g1b, bufA, bufB, Nn);
  // AGG1(bufB) += scatter
  k_spmm<<<spBlocks, 256, 0, stream>>>(er, ec, ev, bufA, bufB, Ee);
  // XW2(bufA) = relu(AGG1) @ g2W^T + b ; zero AGG2(bufB, aliases input rows safely)
  k_linear<true><<<linBlocks, 256, 0, stream>>>(bufB, g2W, g2b, bufA, bufB, Nn);
  // AGG2(bufB) += scatter
  k_spmm<<<spBlocks, 256, 0, stream>>>(er, ec, ev, bufA, bufB, Ee);
  // fused GRU + head, reads relu(AGG2) internally
  k_fused<<<fusedBlocks, 64, 0, stream>>>(Xd, bufB, (const uint*)wpk, bih, bhh,
                                          h1b, h2W, h2b, out, Nn, Tt);
}
